// GATEncoder_6932077215863
// MI455X (gfx1250) — compile-verified, matrix operand order checked by reading, not run on hardware
//
#include <hip/hip_runtime.h>
#include <math.h>

#define LRELU_SLOPE 0.2f

typedef __attribute__((ext_vector_type(2))) float v2f;
typedef __attribute__((ext_vector_type(8))) float v8f;
#if !__has_builtin(__builtin_amdgcn_wmma_f32_16x16x4_f32)
typedef __attribute__((ext_vector_type(16))) __bf16 v16bf;
#endif

static inline unsigned cdiv_u(long a, long b) { return (unsigned)((a + b - 1) / b); }

// -------- float atomic max via int/uint ordering trick (init with -inf) -----
__device__ __forceinline__ void atomicMaxF(float* addr, float val) {
  if (val >= 0.0f) atomicMax((int*)addr, __float_as_int(val));
  else             atomicMin((unsigned int*)addr, __float_as_uint(val));
}

// ---------------------------------------------------------------------------
// GEMM: C[N x NCOLS] = A[N x 128] * B[128 x NCOLS], fp32, WMMA 16x16x4_f32.
// One block = 16-row strip (A strip staged in LDS), one wave per 16-col tile.
// A layout (16x4 f32): VGPR v, lanes 0-15 -> K = kb + 2*(lane>=16) + v, M=lane&15
// B layout (4x16 f32): same K mapping, N = lane&15
// C layout: VGPR v -> row M = v + 8*(lane>=16), N = lane&15
// ---------------------------------------------------------------------------
template<int NCOLS>
__global__ void gemm16_wmma(const float* __restrict__ A,
                            const float* __restrict__ B,
                            float* __restrict__ C) {
  __shared__ float As[16 * 128];
  const int row0 = blockIdx.x << 4;
  for (int i = threadIdx.x; i < 16 * 128; i += blockDim.x)
    As[i] = A[(size_t)(row0 + (i >> 7)) * 128 + (i & 127)];
  __syncthreads();

  const int lane = threadIdx.x & 31;
  const int wv   = threadIdx.x >> 5;
  const int col  = (wv << 4) + (lane & 15);
  const int hi   = lane >> 4;             // 0 or 1 (half-wave)
  const int ar   = (lane & 15) * 128;
  v8f acc = {};

#if __has_builtin(__builtin_amdgcn_wmma_f32_16x16x4_f32)
  const int kh = hi * 2;
  for (int kb = 0; kb < 128; kb += 4) {
    v2f a, b;
    a.x = As[ar + kb + kh];
    a.y = As[ar + kb + kh + 1];
    b.x = B[(size_t)(kb + kh)     * NCOLS + col];
    b.y = B[(size_t)(kb + kh + 1) * NCOLS + col];
    acc = __builtin_amdgcn_wmma_f32_16x16x4_f32(false, a, false, b,
                                                (short)0, acc, false, false);
  }
#else
  // bf16 fallback (codegen-confirmed builtin); 16-bit 16x32 A layout
  for (int kb = 0; kb < 128; kb += 32) {
    v16bf a, b;
#pragma unroll
    for (int e = 0; e < 16; ++e) {
      const int vg = e >> 1;
      const int k  = kb + ((vg & 3) * 2 + (e & 1)) + 8 * hi + 16 * (vg >> 2);
      a[e] = (__bf16)As[ar + k];
      b[e] = (__bf16)B[(size_t)k * NCOLS + col];
    }
    acc = __builtin_amdgcn_wmma_f32_16x16x32_bf16(false, a, false, b,
                                                  (short)0, acc, false, false);
  }
#endif

#pragma unroll
  for (int v = 0; v < 8; ++v)
    C[(size_t)(row0 + v + (hi << 3)) * NCOLS + col] = acc[v];
}

// ---------------------------------------------------------------------------
__global__ void fill_f32(float* __restrict__ p, float v, int n) {
  int i = blockIdx.x * blockDim.x + threadIdx.x;
  if (i < n) p[i] = v;
}

// per-(node,head) attention coefficients: als = <h, a_s>, ald = <h, a_d>
__global__ void attn_coef(const float* __restrict__ hf,
                          const float* __restrict__ avs,
                          const float* __restrict__ avd,
                          float* __restrict__ als, float* __restrict__ ald,
                          int N, int H, int HD) {
  int t = blockIdx.x * blockDim.x + threadIdx.x;
  if (t >= N * H) return;
  int n = t / H, h = t - n * H;
  const float* row = hf + (size_t)n * (H * HD) + h * HD;
  float s = 0.f, d = 0.f;
  for (int k = 0; k < HD; ++k) {
    s += row[k] * avs[h * HD + k];
    d += row[k] * avd[h * HD + k];
  }
  als[t] = s; ald[t] = d;
}

__device__ __forceinline__ void edge_nodes(const int* __restrict__ src,
                                           const int* __restrict__ dst,
                                           int E, int e, int& s, int& d) {
  if (e < E) { s = src[e]; d = dst[e]; } else { s = d = e - E; }  // self loops
}

__global__ void edge_max(const int* __restrict__ src, const int* __restrict__ dst,
                         int E, int Etot,
                         const float* __restrict__ als, const float* __restrict__ ald,
                         float* __restrict__ m, int H) {
  int e = blockIdx.x * blockDim.x + threadIdx.x;
  if (e >= Etot) return;
  int s, d; edge_nodes(src, dst, E, e, s, d);
  for (int h = 0; h < H; ++h) {
    float v = als[s * H + h] + ald[d * H + h];
    v = v > 0.f ? v : LRELU_SLOPE * v;
    atomicMaxF(&m[d * H + h], v);
  }
}

__global__ void edge_expsum(const int* __restrict__ src, const int* __restrict__ dst,
                            int E, int Etot,
                            const float* __restrict__ als, const float* __restrict__ ald,
                            const float* __restrict__ m, float* __restrict__ ssum, int H) {
  int e = blockIdx.x * blockDim.x + threadIdx.x;
  if (e >= Etot) return;
  int s, d; edge_nodes(src, dst, E, e, s, d);
  for (int h = 0; h < H; ++h) {
    float v = als[s * H + h] + ald[d * H + h];
    v = v > 0.f ? v : LRELU_SLOPE * v;
    atomicAdd(&ssum[d * H + h], expf(v - m[d * H + h]));
  }
}

// one thread per (edge, 16-feature chunk): alpha * h[src] scatter-add to agg[dst]
__global__ void edge_scatter(const int* __restrict__ src, const int* __restrict__ dst,
                             int E, int Etot,
                             const float* __restrict__ als, const float* __restrict__ ald,
                             const float* __restrict__ m, const float* __restrict__ ssum,
                             const float* __restrict__ hf, float* __restrict__ agg,
                             int H, int HD, int D) {
  const int chunks = D >> 4;
  int t = blockIdx.x * blockDim.x + threadIdx.x;
  if (t >= Etot * chunks) return;
  int e = t / chunks, sub = t - e * chunks;
  int s, d; edge_nodes(src, dst, E, e, s, d);
  const int k0 = sub << 4;
  const int h  = k0 / HD;
  float v = als[s * H + h] + ald[d * H + h];
  v = v > 0.f ? v : LRELU_SLOPE * v;
  const float alpha = expf(v - m[d * H + h]) / (ssum[d * H + h] + 1e-16f);
  const float* hrow = hf  + (size_t)s * D + k0;
  float*       arow = agg + (size_t)d * D + k0;
#pragma unroll
  for (int k = 0; k < 16; ++k)
    atomicAdd(&arow[k], alpha * hrow[k]);
}

// BatchNorm column stats: blockDim.x == 128 (one thread per column)
#define BN_ROWS 512
__global__ void bn_stats(const float* __restrict__ v, float* __restrict__ stats, int N) {
  const int col = threadIdx.x;
  int r0 = blockIdx.x * BN_ROWS;
  int r1 = r0 + BN_ROWS; if (r1 > N) r1 = N;
  float s = 0.f, q = 0.f;
  for (int r = r0; r < r1; ++r) {
    float x = v[(size_t)r * 128 + col];
    s += x; q += x * x;
  }
  atomicAdd(&stats[col], s);
  atomicAdd(&stats[128 + col], q);
}

// relu(BN(agg)) + residual -> xcur (in place). Bias b cancels under BN.
__global__ void bn_apply(const float* __restrict__ agg, const float* __restrict__ stats,
                         const float* __restrict__ g, const float* __restrict__ bt,
                         float* __restrict__ xcur, int N) {
  int t = blockIdx.x * blockDim.x + threadIdx.x;
  if (t >= N * 128) return;
  const int j = t & 127;
  const float invN = 1.0f / (float)N;
  const float mu  = stats[j] * invN;
  const float var = stats[128 + j] * invN - mu * mu;
  float y = (agg[t] - mu) * rsqrtf(var + 1e-5f) * g[j] + bt[j];
  y = y > 0.f ? y : 0.f;
  xcur[t] = y + xcur[t];
}

// final: out = LayerNorm(agg + b2) * g + b ; one wave32 per node (64 feats)
__global__ void ln_out(const float* __restrict__ agg, const float* __restrict__ b2,
                       const float* __restrict__ g, const float* __restrict__ bt,
                       float* __restrict__ out, int N) {
  int wid  = (blockIdx.x * blockDim.x + threadIdx.x) >> 5;
  int lane = threadIdx.x & 31;
  if (wid >= N) return;
  float x0 = agg[(size_t)wid * 64 + lane]      + b2[lane];
  float x1 = agg[(size_t)wid * 64 + 32 + lane] + b2[32 + lane];
  float s = x0 + x1, q = x0 * x0 + x1 * x1;
  for (int o = 16; o > 0; o >>= 1) {
    s += __shfl_xor(s, o, 32);
    q += __shfl_xor(q, o, 32);
  }
  const float mu  = s * (1.0f / 64.0f);
  const float var = q * (1.0f / 64.0f) - mu * mu;
  const float r   = rsqrtf(var + 1e-5f);
  out[(size_t)wid * 64 + lane]      = (x0 - mu) * r * g[lane]      + bt[lane];
  out[(size_t)wid * 64 + 32 + lane] = (x1 - mu) * r * g[32 + lane] + bt[32 + lane];
}

// ---------------------------------------------------------------------------
extern "C" void kernel_launch(void* const* d_in, const int* in_sizes, int n_in,
                              void* d_out, int out_size, void* d_ws, size_t ws_size,
                              hipStream_t stream) {
  const float* x   = (const float*)d_in[0];
  const int*   ei  = (const int*)  d_in[1];
  const float* W0  = (const float*)d_in[2];
  const float* as0 = (const float*)d_in[3];
  const float* ad0 = (const float*)d_in[4];
  // d_in[5] = b0 (cancels under BN)
  const float* g0  = (const float*)d_in[6];
  const float* bt0 = (const float*)d_in[7];
  const float* W1  = (const float*)d_in[8];
  const float* as1 = (const float*)d_in[9];
  const float* ad1 = (const float*)d_in[10];
  const float* g1  = (const float*)d_in[12];
  const float* bt1 = (const float*)d_in[13];
  const float* W2  = (const float*)d_in[14];
  const float* as2 = (const float*)d_in[15];
  const float* ad2 = (const float*)d_in[16];
  const float* b2  = (const float*)d_in[17];
  const float* lng = (const float*)d_in[18];
  const float* lnb = (const float*)d_in[19];

  const int N    = in_sizes[0] / 128;    // 100000 (multiple of 16)
  const int E    = in_sizes[1] / 2;      // 800000
  const int Etot = E + N;
  const int* srcIdx = ei;
  const int* dstIdx = ei + E;

  float* ws   = (float*)d_ws;
  float* xcur = ws;                          // N*128
  float* h    = xcur + (size_t)N * 128;      // N*128
  float* agg  = h    + (size_t)N * 128;      // N*128
  float* als  = agg  + (size_t)N * 128;      // N*8
  float* ald  = als  + (size_t)N * 8;        // N*8
  float* mseg = ald  + (size_t)N * 8;        // N*8
  float* ssum = mseg + (size_t)N * 8;        // N*8
  float* stats= ssum + (size_t)N * 8;        // 256

  hipMemcpyAsync(xcur, x, (size_t)N * 128 * sizeof(float),
                 hipMemcpyDeviceToDevice, stream);

  const int T = 256;
  auto hidden_layer = [&](const float* W, const float* as, const float* ad,
                          const float* g, const float* bt) {
    const int H = 8, HD = 16, D = 128;
    gemm16_wmma<128><<<N / 16, 256, 0, stream>>>(xcur, W, h);
    fill_f32<<<cdiv_u((long)N * H, T), T, 0, stream>>>(mseg, -INFINITY, N * H);
    fill_f32<<<cdiv_u((long)N * H, T), T, 0, stream>>>(ssum, 0.f, N * H);
    fill_f32<<<1, 256, 0, stream>>>(stats, 0.f, 256);
    fill_f32<<<cdiv_u((long)N * D, T), T, 0, stream>>>(agg, 0.f, N * D);
    attn_coef<<<cdiv_u((long)N * H, T), T, 0, stream>>>(h, as, ad, als, ald, N, H, HD);
    edge_max   <<<cdiv_u(Etot, T), T, 0, stream>>>(srcIdx, dstIdx, E, Etot, als, ald, mseg, H);
    edge_expsum<<<cdiv_u(Etot, T), T, 0, stream>>>(srcIdx, dstIdx, E, Etot, als, ald, mseg, ssum, H);
    edge_scatter<<<cdiv_u((long)Etot * (D / 16), T), T, 0, stream>>>(
        srcIdx, dstIdx, E, Etot, als, ald, mseg, ssum, h, agg, H, HD, D);
    bn_stats<<<cdiv_u(N, BN_ROWS), 128, 0, stream>>>(agg, stats, N);
    bn_apply<<<cdiv_u((long)N * 128, T), T, 0, stream>>>(agg, stats, g, bt, xcur, N);
  };

  hidden_layer(W0, as0, ad0, g0, bt0);
  hidden_layer(W1, as1, ad1, g1, bt1);

  // final layer: 1 head x 64, concat=False, then bias + LayerNorm -> d_out
  {
    const int H = 1, HD = 64, D = 64;
    gemm16_wmma<64><<<N / 16, 128, 0, stream>>>(xcur, W2, h);   // h used as [N,64]
    fill_f32<<<cdiv_u(N, T), T, 0, stream>>>(mseg, -INFINITY, N);
    fill_f32<<<cdiv_u(N, T), T, 0, stream>>>(ssum, 0.f, N);
    fill_f32<<<cdiv_u((long)N * D, T), T, 0, stream>>>(agg, 0.f, N * D);
    attn_coef<<<cdiv_u(N, T), T, 0, stream>>>(h, as2, ad2, als, ald, N, H, HD);
    edge_max   <<<cdiv_u(Etot, T), T, 0, stream>>>(srcIdx, dstIdx, E, Etot, als, ald, mseg, H);
    edge_expsum<<<cdiv_u(Etot, T), T, 0, stream>>>(srcIdx, dstIdx, E, Etot, als, ald, mseg, ssum, H);
    edge_scatter<<<cdiv_u((long)Etot * (D / 16), T), T, 0, stream>>>(
        srcIdx, dstIdx, E, Etot, als, ald, mseg, ssum, h, agg, H, HD, D);
    ln_out<<<cdiv_u((long)N * 32, T), T, 0, stream>>>(agg, b2, lng, lnb, (float*)d_out, N);
  }
}
